// LSTM_Layer_48601849922172
// MI455X (gfx1250) — matrix operand order repaired and checked by reading, not compiled, not run on hardware
//
#include <hip/hip_runtime.h>
#include <stdint.h>

// Problem dims (fixed by the reference)
#define B_DIM 32
#define T_DIM 512
#define I_DIM 1024
#define H_DIM 1024
#define G_DIM 4096            // 4*H
#define BT    (B_DIM * T_DIM) // 16384
#define NWG_R 64              // H_DIM / 16 column-slices for the recurrent kernel

typedef __bf16 bf16;
typedef __attribute__((ext_vector_type(8)))  float v8f;
typedef __attribute__((ext_vector_type(8)))  bf16  v8bf;
typedef __attribute__((ext_vector_type(16))) bf16  v16bf;
typedef int v4i __attribute__((vector_size(16)));  // matches builtin's param type

// ---------------------------------------------------------------------------
// CDNA5 async global->LDS copy (ASYNCcnt path), guarded so the file compiles
// on toolchains that lack the builtin (falls back to plain ld/st staging).
// Builtin signature (from hipcc diagnostic): param0 = v4i in AS1 (global),
// param1 = LDS dest, then imm offset + cpol.
// ---------------------------------------------------------------------------
#if __has_builtin(__builtin_amdgcn_global_load_async_to_lds_b128)
#define HAS_ASYNC_LDS 1
#endif

__device__ __forceinline__ void cp16_g2l(void* lds_dst, const void* gsrc) {
#if defined(HAS_ASYNC_LDS)
  __builtin_amdgcn_global_load_async_to_lds_b128(
      (__attribute__((address_space(1))) v4i*)gsrc,
      (__attribute__((address_space(3))) v4i*)lds_dst,
      /*imm offset*/ 0, /*cpol*/ 0);
#else
  struct alignas(16) B16 { uint32_t u[4]; };
  *(B16*)lds_dst = *(const B16*)gsrc;
#endif
}

__device__ __forceinline__ void async_fence() {
#if defined(HAS_ASYNC_LDS)
#if __has_builtin(__builtin_amdgcn_s_wait_asynccnt)
  __builtin_amdgcn_s_wait_asynccnt(0);
#else
  asm volatile("s_wait_asynccnt 0" ::: "memory");
#endif
#endif
}

// ---------------------------------------------------------------------------
// WMMA fragment loaders for V_WMMA_F32_16X16X32_BF16 (wave32).
// A 16x32 (MxK): lane m=lane&15; lanes<16 hold K=0..15, lanes>=16 hold
// K=16..31  -> two contiguous 16B runs per lane.
// B 32x16 (KxN) stored N-major (tileT[n*ld + k]): lane n=lane&15; lanes<16 hold
// K=0..15, lanes>=16 hold K=16..31 -> one contiguous 32B run per lane.
// C/D 16x16 f32: lane n=lane&15; vgpr r -> m = r + (lane<16 ? 0 : 8).
// ---------------------------------------------------------------------------
__device__ __forceinline__ v16bf frag_A(const bf16* tile, int ld, int m0) {
  const int lane = (int)(threadIdx.x & 31);
  const bf16* p = tile + (size_t)(m0 + (lane & 15)) * ld + ((lane < 16) ? 0 : 16);
  v8bf lo = *(const v8bf*)p;
  v8bf hi = *(const v8bf*)(p + 8);
  v16bf r;
#pragma unroll
  for (int e = 0; e < 8; ++e) { r[e] = lo[e]; r[e + 8] = hi[e]; }
  return r;
}

__device__ __forceinline__ v16bf frag_B(const bf16* tileT, int ld, int n0) {
  const int lane = (int)(threadIdx.x & 31);
  const bf16* p = tileT + (size_t)(n0 + (lane & 15)) * ld + ((lane < 16) ? 0 : 16);
  return *(const v16bf*)p;
}

__device__ __forceinline__ v8f wmma_bf16(v16bf a, v16bf b, v8f c) {
  return __builtin_amdgcn_wmma_f32_16x16x32_bf16(false, a, false, b, (short)0, c,
                                                 false, false);
}

// ---------------------------------------------------------------------------
// fp32 -> bf16 conversion (grid-stride)
// ---------------------------------------------------------------------------
__global__ void cvt_bf16(const float* __restrict__ src, bf16* __restrict__ dst,
                         int n) {
  int i = blockIdx.x * blockDim.x + threadIdx.x;
  int stride = gridDim.x * blockDim.x;
  for (; i < n; i += stride) dst[i] = (bf16)src[i];
}

// ---------------------------------------------------------------------------
// wx = x @ W^T + b_w   (bf16 in, f32 accum, bf16 out)
// Block tile 128x128, 8 waves (2x4), wave tile 64x32 = 4x2 WMMA tiles, K=32/step.
// A tile: As[row][k] (128x32), B tile: Bs[n][k] (128x32, i.e. W rows verbatim).
// Double-buffered async LDS staging.
// ---------------------------------------------------------------------------
__global__ __launch_bounds__(256) void wx_gemm(const bf16* __restrict__ xb,
                                               const bf16* __restrict__ wb,
                                               const float* __restrict__ bw,
                                               bf16* __restrict__ wx) {
  __shared__ alignas(32) bf16 As[2][128 * 32];
  __shared__ alignas(32) bf16 Bs[2][128 * 32];

  const int tid = (int)threadIdx.x;
  const int lane = tid & 31;
  const int wave = tid >> 5;
  const int mw = wave >> 2;  // 0..1 -> 64 rows each
  const int nw = wave & 3;   // 0..3 -> 32 cols each
  const int blockM = (int)blockIdx.x * 128;
  const int blockN = (int)blockIdx.y * 128;

  v8f acc[4][2] = {};

  // stage k-chunk kc into buffer `buf`
  auto stage = [&](int buf, int kc) {
#pragma unroll
    for (int it = 0; it < 2; ++it) {
      const int slot = tid + it * 256;  // 0..511
      const int row = slot >> 2;        // 0..127
      const int seg = slot & 3;         // 4 x 16B per 64B row
      cp16_g2l(&As[buf][row * 32 + seg * 8],
               xb + (size_t)(blockM + row) * I_DIM + kc + seg * 8);
      cp16_g2l(&Bs[buf][row * 32 + seg * 8],
               wb + (size_t)(blockN + row) * I_DIM + kc + seg * 8);
    }
  };

  stage(0, 0);
  for (int kc = 0; kc < I_DIM; kc += 32) {
    const int buf = (kc >> 5) & 1;
    async_fence();
    __syncthreads();
    if (kc + 32 < I_DIM) stage(buf ^ 1, kc + 32);
    if (kc + 64 < I_DIM) {  // L2 prefetch two chunks ahead
      __builtin_prefetch(xb + (size_t)(blockM + (tid >> 1)) * I_DIM + kc + 64, 0, 0);
      __builtin_prefetch(wb + (size_t)(blockN + (tid >> 1)) * I_DIM + kc + 64, 0, 0);
    }
    v16bf af[4];
#pragma unroll
    for (int m = 0; m < 4; ++m) af[m] = frag_A(&As[buf][0], 32, mw * 64 + m * 16);
    v16bf bf_[2];
#pragma unroll
    for (int n = 0; n < 2; ++n) bf_[n] = frag_B(&Bs[buf][0], 32, nw * 32 + n * 16);
#pragma unroll
    for (int m = 0; m < 4; ++m)
#pragma unroll
      for (int n = 0; n < 2; ++n) acc[m][n] = wmma_bf16(af[m], bf_[n], acc[m][n]);
  }

  const int rbase = (lane < 16) ? 0 : 8;
#pragma unroll
  for (int m = 0; m < 4; ++m)
#pragma unroll
    for (int n = 0; n < 2; ++n) {
      const int gn = blockN + nw * 32 + n * 16 + (lane & 15);
      const float bias = bw[gn];
#pragma unroll
      for (int r = 0; r < 8; ++r) {
        const int gm = blockM + mw * 64 + m * 16 + r + rbase;
        wx[(size_t)gm * G_DIM + gn] = (bf16)(acc[m][n][r] + bias);
      }
    }
}

// ---------------------------------------------------------------------------
// Persistent recurrent scan. 64 WGs x 64 threads (2 waves). WG w owns h-cols
// [16w,16w+16) and the 4 matching gate strips. Per step: gates = wx_t +
// h @ U^T + b_u via WMMA (K=1024 in 32-chunks, U tile double-buffered in LDS),
// then lane-local gate math, then device-wide barrier (monotonic atomic).
// h is double-buffered in global so only one barrier per step is needed.
// ---------------------------------------------------------------------------
__device__ __forceinline__ float sigf(float x) { return 1.0f / (1.0f + __expf(-x)); }
__device__ __forceinline__ float tanh_fast(float x) {
  const float e = __expf(2.0f * x);
  return (e - 1.0f) / (e + 1.0f);
}

__device__ __forceinline__ void grid_barrier(unsigned* bar, unsigned target) {
  __syncthreads();
  if (threadIdx.x == 0) {
    atomicAdd(bar, 1u);
    while (atomicAdd(bar, 0u) < target) __builtin_amdgcn_s_sleep(2);
  }
  __syncthreads();
  __threadfence();  // acquire side
}

__global__ __launch_bounds__(64) void lstm_scan(
    const bf16* __restrict__ wx, const bf16* __restrict__ ub,
    const float* __restrict__ bu, bf16* __restrict__ hb,  // 2 x [32 x 1024]
    float* __restrict__ cbuf, float* __restrict__ hs, float* __restrict__ cs,
    unsigned* __restrict__ bar) {
  __shared__ alignas(32) bf16 Bs[2][64 * 32];  // [n(4 gates x 16)][k]

  const int tid = (int)threadIdx.x;
  const int lane = tid & 31;
  const int wave = tid >> 5;    // 2 waves: M-tiles 0 / 1 (batch rows 0-15 / 16-31)
  const int m0 = wave * 16;
  const int hcol0 = (int)blockIdx.x * 16;
  const int nl = lane & 15;
  const int nH = hcol0 + nl;
  const int rbase = (lane < 16) ? 0 : 8;
  const float bu0 = bu[0 * H_DIM + nH], bu1 = bu[1 * H_DIM + nH];
  const float bu2 = bu[2 * H_DIM + nH], bu3 = bu[3 * H_DIM + nH];

  auto stageU = [&](int buf, int kc) {
#pragma unroll
    for (int it = 0; it < 4; ++it) {
      const int slot = tid + it * 64;  // 0..255
      const int j = slot >> 2;         // 0..63 = g*16 + n
      const int seg = slot & 3;        // 4 x 16B per 64B row
      const int g = j >> 4, n = j & 15;
      cp16_g2l(&Bs[buf][j * 32 + seg * 8],
               ub + (size_t)(g * H_DIM + hcol0 + n) * H_DIM + kc + seg * 8);
    }
  };

  for (int t = 0; t < T_DIM; ++t) {
    const bf16* hbr = hb + (size_t)(t & 1) * (B_DIM * H_DIM);
    bf16* hbw = hb + (size_t)((t + 1) & 1) * (B_DIM * H_DIM);

    v8f acc[4] = {};
    stageU(0, 0);
    for (int kc = 0; kc < H_DIM; kc += 32) {
      const int buf = (kc >> 5) & 1;
      async_fence();
      __syncthreads();
      if (kc + 32 < H_DIM) stageU(buf ^ 1, kc + 32);
      const v16bf a = frag_A(hbr + kc, H_DIM, m0);  // h stays L2-hot; read direct
#pragma unroll
      for (int g = 0; g < 4; ++g) {
        const v16bf b = frag_B(&Bs[buf][0], 32, g * 16);
        acc[g] = wmma_bf16(a, b, acc[g]);
      }
    }

    // Gate math: acc[0..3] = i,f,g,o at the SAME (m, nH) per lane element.
#pragma unroll
    for (int r = 0; r < 8; ++r) {
      const int m = m0 + r + rbase;  // batch row 0..31
      const size_t wrow = ((size_t)m * T_DIM + t) * G_DIM;
      const float gi = acc[0][r] + (float)wx[wrow + 0 * H_DIM + nH] + bu0;
      const float gf = acc[1][r] + (float)wx[wrow + 1 * H_DIM + nH] + bu1;
      const float gg = acc[2][r] + (float)wx[wrow + 2 * H_DIM + nH] + bu2;
      const float go = acc[3][r] + (float)wx[wrow + 3 * H_DIM + nH] + bu3;
      const int sidx = m * H_DIM + nH;
      const float cold = cbuf[sidx];
      const float cnew = sigf(gf) * cold + sigf(gi) * tanh_fast(gg);
      const float hnew = sigf(go) * fmaxf(cnew, 0.0f);
      cbuf[sidx] = cnew;
      hbw[sidx] = (bf16)hnew;
      const size_t oidx = (size_t)m * (T_DIM * H_DIM) + (size_t)t * H_DIM + nH;
      hs[oidx] = hnew;
      cs[oidx] = cnew;
    }
    __threadfence();  // release side
    grid_barrier(bar, (unsigned)(t + 1) * NWG_R);
  }
}

// ---------------------------------------------------------------------------
// Host launcher. Workspace layout (needs ~177 MB of d_ws):
//   wx   bf16 [BT,4H]    128 MB
//   xb   bf16 [BT,I]      32 MB
//   wb   bf16 [4H,I]       8 MB
//   ub   bf16 [4H,H]       8 MB
//   hb   bf16 2x[32,H]   128 KB   (zeroed each launch)
//   cbuf f32  [32,H]     128 KB   (zeroed each launch)
//   bar  u32             256 B    (zeroed each launch)
// ---------------------------------------------------------------------------
extern "C" void kernel_launch(void* const* d_in, const int* in_sizes, int n_in,
                              void* d_out, int out_size, void* d_ws, size_t ws_size,
                              hipStream_t stream) {
  (void)in_sizes; (void)n_in; (void)out_size; (void)ws_size;
  const float* x  = (const float*)d_in[0];
  const float* W  = (const float*)d_in[1];
  const float* bw = (const float*)d_in[2];
  const float* U  = (const float*)d_in[3];
  const float* bu = (const float*)d_in[4];
  float* hs = (float*)d_out;
  float* cs = hs + (size_t)B_DIM * T_DIM * H_DIM;

  char* ws = (char*)d_ws;
  size_t off = 0;
  bf16* wx = (bf16*)(ws + off); off += (size_t)BT * G_DIM * sizeof(bf16);
  bf16* xb = (bf16*)(ws + off); off += (size_t)BT * I_DIM * sizeof(bf16);
  bf16* wb = (bf16*)(ws + off); off += (size_t)G_DIM * I_DIM * sizeof(bf16);
  bf16* ub = (bf16*)(ws + off); off += (size_t)G_DIM * H_DIM * sizeof(bf16);
  const size_t state_off = off;
  bf16* hbuf = (bf16*)(ws + off); off += (size_t)2 * B_DIM * H_DIM * sizeof(bf16);
  float* cbuf = (float*)(ws + off); off += (size_t)B_DIM * H_DIM * sizeof(float);
  unsigned* bar = (unsigned*)(ws + off); off += 256;

  // Deterministic per-call init of h0/c0/barrier (graph-capture safe).
  (void)hipMemsetAsync(ws + state_off, 0, off - state_off, stream);

  cvt_bf16<<<2048, 256, 0, stream>>>(x, xb, BT * I_DIM);
  cvt_bf16<<<1024, 256, 0, stream>>>(W, wb, G_DIM * I_DIM);
  cvt_bf16<<<1024, 256, 0, stream>>>(U, ub, G_DIM * H_DIM);
  wx_gemm<<<dim3(BT / 128, G_DIM / 128), 256, 0, stream>>>(xb, wb, bw, wx);
  lstm_scan<<<NWG_R, 64, 0, stream>>>(wx, ub, bu, hbuf, cbuf, hs, cs, bar);
}
